// BinarizedCIFARNetwork_35527969472951
// MI455X (gfx1250) — 8-bit matrix operand order repaired and checked by reading, compiled, not run on hardware
//
#include <hip/hip_runtime.h>
#include <stdint.h>

typedef int v8i __attribute__((ext_vector_type(8)));
typedef int v4ig __attribute__((vector_size(16)));

union V8U {
    v8i v;
    unsigned long long u[4];
    int4 q[2];
};

#define EPSV 1e-5f

#if defined(__has_builtin)
#if __has_builtin(__builtin_amdgcn_global_load_async_to_lds_b128)
#define HAVE_ASYNC_LDS 1
#endif
#if __has_builtin(__builtin_amdgcn_s_wait_asynccnt)
#define HAVE_WAIT_ASYNC 1
#endif
#endif

__device__ __forceinline__ void stage16(signed char* lds_dst, const signed char* gsrc)
{
#ifdef HAVE_ASYNC_LDS
    __builtin_amdgcn_global_load_async_to_lds_b128(
        (__attribute__((address_space(1))) v4ig*)(size_t)gsrc,
        (__attribute__((address_space(3))) v4ig*)(unsigned)(size_t)lds_dst,
        0, 0);
#else
    *(int4*)lds_dst = *(const int4*)gsrc;
#endif
}

__device__ __forceinline__ void async_fence()
{
#ifdef HAVE_ASYNC_LDS
#ifdef HAVE_WAIT_ASYNC
    __builtin_amdgcn_s_wait_asynccnt(0);
#else
    asm volatile("s_wait_asynccnt 0x0" ::: "memory");
#endif
#endif
}

__global__ void conv0_direct(const float* __restrict__ x,
                             const float* __restrict__ w0,
                             const float* __restrict__ b0,
                             float* __restrict__ out,
                             int NB)
{
    int idx = blockIdx.x * blockDim.x + threadIdx.x;
    int total = NB * 32 * 32 * 128;
    if (idx >= total) return;
    int o = idx & 127;
    int t = idx >> 7;
    int w = t & 31; t >>= 5;
    int h = t & 31; t >>= 5;
    int n = t;
    float acc = b0[o];
    for (int c = 0; c < 3; ++c)
        for (int dy = 0; dy < 3; ++dy) {
            int hh = h + dy - 1;
            if (hh < 0 || hh >= 32) continue;
            for (int dx = 0; dx < 3; ++dx) {
                int ww = w + dx - 1;
                if (ww < 0 || ww >= 32) continue;
                float wv = w0[((o * 3 + c) * 3 + dy) * 3 + dx];
                float sw = (wv > 0.f) ? 1.f : ((wv < 0.f) ? -1.f : 0.f);
                acc += sw * x[((n * 3 + c) * 32 + hh) * 32 + ww];
            }
        }
    out[idx] = acc;
}

__global__ void pack_conv_w(const float* __restrict__ w,
                            signed char* __restrict__ wp,
                            int C, int O)
{
    int idx = blockIdx.x * blockDim.x + threadIdx.x;
    int Nt = O >> 4, K64 = C >> 6;
    int total = 9 * K64 * Nt * 1024;
    if (idx >= total) return;
    int j = idx & 31;
    int lane = (idx >> 5) & 31;
    int t = idx >> 10;
    int nc = t % Nt; t /= Nt;
    int kc = t % K64;
    int tap = t / K64;
    int n = nc * 16 + (lane & 15);
    int khalf = lane >> 4;
    int k = kc * 64 + 16 * (j >> 3) + khalf * 8 + (j & 7);
    float wv = w[(n * C + k) * 9 + tap];
    wp[idx] = (wv > 0.f) ? (signed char)1 : ((wv < 0.f) ? (signed char)-1 : (signed char)0);
}

__global__ void pack_fc_w(const float* __restrict__ w,
                          signed char* __restrict__ wp,
                          int K, int N, int chw_remap)
{
    int idx = blockIdx.x * blockDim.x + threadIdx.x;
    int Nt = N >> 4, K64 = K >> 6;
    int total = K64 * Nt * 1024;
    if (idx >= total) return;
    int j = idx & 31;
    int lane = (idx >> 5) & 31;
    int t = idx >> 10;
    int nc = t % Nt;
    int kc = t / Nt;
    int n = nc * 16 + (lane & 15);
    int khalf = lane >> 4;
    int k = kc * 64 + 16 * (j >> 3) + khalf * 8 + (j & 7);
    int src;
    if (chw_remap) {
        int c = k & 511;
        int s = k >> 9;
        src = n * K + c * 16 + s;
    } else {
        src = n * K + k;
    }
    float wv = w[src];
    wp[idx] = (wv > 0.f) ? (signed char)1 : ((wv < 0.f) ? (signed char)-1 : (signed char)0);
}

#define WMMA_IU8(A, B, C) \
    __builtin_amdgcn_wmma_i32_16x16x64_iu8(true, (A), true, (B), (C), false, false)

__global__ void __launch_bounds__(256)
bgemm_conv(const signed char* __restrict__ act,
           const signed char* __restrict__ wp,
           const float* __restrict__ bias,
           float* __restrict__ out,
           int NB, int H, int W, int Cin, int Cout)
{
    __shared__ signed char Bsh[2][4096];
    int tid = threadIdx.x;
    int lane = tid & 31;
    int wave = tid >> 5;

    int Nt = Cout >> 4, Nt4 = Cout >> 6;
    int tn4 = blockIdx.x % Nt4;
    int tmblk = blockIdx.x / Nt4;
    int tm = (tmblk << 3) + wave;
    int m0 = tm << 4;

    int K64 = Cin >> 6;
    int T9 = 9 * K64;
    long ldw = (long)Nt * 1024;
    const signed char* wsrc = wp + (long)tn4 * 4096 + tid * 16;

    int mrow = m0 + (lane & 15);
    int wq = mrow % W;
    int t = mrow / W;
    int hq = t % H;
    int nimg = t / H;
    int kbase = (lane < 16) ? 0 : 8;

    v8i acc0 = {0, 0, 0, 0, 0, 0, 0, 0};
    v8i acc1 = acc0, acc2 = acc0, acc3 = acc0;

    stage16(&Bsh[0][tid * 16], wsrc);
    int buf = 0;
    int it = 0;
    for (int tap = 0; tap < 9; ++tap) {
        int dy = tap / 3 - 1, dx = tap % 3 - 1;
        int h2 = hq + dy, w2 = wq + dx;
        bool valid = (h2 >= 0) && (h2 < H) && (w2 >= 0) && (w2 < W);
        const signed char* ap =
            act + ((long)(nimg * H + h2) * W + w2) * Cin + kbase;
        V8U a;
        a.u[0] = 0; a.u[1] = 0; a.u[2] = 0; a.u[3] = 0;
        for (int kc = 0; kc < K64; ++kc, ++it) {
            async_fence();
            __syncthreads();
            if (it + 1 < T9)
                stage16(&Bsh[buf ^ 1][tid * 16], wsrc + (long)(it + 1) * ldw);
            if (valid) {
                const unsigned long long* q =
                    (const unsigned long long*)(ap + kc * 64);
                a.u[0] = q[0]; a.u[1] = q[2]; a.u[2] = q[4]; a.u[3] = q[6];
            }
            const signed char* bb = &Bsh[buf][lane * 32];
            V8U b0, b1, b2, b3;
            b0.q[0] = ((const int4*)bb)[0];
            b0.q[1] = ((const int4*)bb)[1];
            b1.q[0] = ((const int4*)(bb + 1024))[0];
            b1.q[1] = ((const int4*)(bb + 1024))[1];
            b2.q[0] = ((const int4*)(bb + 2048))[0];
            b2.q[1] = ((const int4*)(bb + 2048))[1];
            b3.q[0] = ((const int4*)(bb + 3072))[0];
            b3.q[1] = ((const int4*)(bb + 3072))[1];
            acc0 = WMMA_IU8(a.v, b0.v, acc0);
            acc1 = WMMA_IU8(a.v, b1.v, acc1);
            acc2 = WMMA_IU8(a.v, b2.v, acc2);
            acc3 = WMMA_IU8(a.v, b3.v, acc3);
            buf ^= 1;
        }
    }

    int ncol = (tn4 << 6) + (lane & 15);
    int mh = (lane < 16) ? 0 : 8;
    float bv0 = bias[ncol], bv1 = bias[ncol + 16];
    float bv2 = bias[ncol + 32], bv3 = bias[ncol + 48];
    for (int r = 0; r < 8; ++r) {
        long row = (long)(m0 + r + mh) * Cout;
        out[row + ncol]      = (float)acc0[r] + bv0;
        out[row + ncol + 16] = (float)acc1[r] + bv1;
        out[row + ncol + 32] = (float)acc2[r] + bv2;
        out[row + ncol + 48] = (float)acc3[r] + bv3;
    }
}

__global__ void __launch_bounds__(256)
bgemm_fc(const signed char* __restrict__ act,
         const signed char* __restrict__ wp,
         const float* __restrict__ bias,
         float* __restrict__ out,
         int K, int N)
{
    __shared__ signed char Bsh[2][4096];
    int tid = threadIdx.x;
    int lane = tid & 31;
    int wave = tid >> 5;

    int Nt = N >> 4;
    int tn4 = blockIdx.x;
    int m0 = wave << 4;

    int K64 = K >> 6;
    long ldw = (long)Nt * 1024;
    const signed char* wsrc = wp + (long)tn4 * 4096 + tid * 16;

    int mrow = m0 + (lane & 15);
    int kbase = (lane < 16) ? 0 : 8;
    long abase = (long)mrow * K + kbase;

    v8i acc0 = {0, 0, 0, 0, 0, 0, 0, 0};
    v8i acc1 = acc0, acc2 = acc0, acc3 = acc0;

    stage16(&Bsh[0][tid * 16], wsrc);
    int buf = 0;
    for (int kc = 0; kc < K64; ++kc) {
        async_fence();
        __syncthreads();
        if (kc + 1 < K64)
            stage16(&Bsh[buf ^ 1][tid * 16], wsrc + (long)(kc + 1) * ldw);
        V8U a;
        const unsigned long long* q =
            (const unsigned long long*)(act + abase + kc * 64);
        a.u[0] = q[0]; a.u[1] = q[2]; a.u[2] = q[4]; a.u[3] = q[6];
        const signed char* bb = &Bsh[buf][lane * 32];
        V8U b0, b1, b2, b3;
        b0.q[0] = ((const int4*)bb)[0];
        b0.q[1] = ((const int4*)bb)[1];
        b1.q[0] = ((const int4*)(bb + 1024))[0];
        b1.q[1] = ((const int4*)(bb + 1024))[1];
        b2.q[0] = ((const int4*)(bb + 2048))[0];
        b2.q[1] = ((const int4*)(bb + 2048))[1];
        b3.q[0] = ((const int4*)(bb + 3072))[0];
        b3.q[1] = ((const int4*)(bb + 3072))[1];
        acc0 = WMMA_IU8(a.v, b0.v, acc0);
        acc1 = WMMA_IU8(a.v, b1.v, acc1);
        acc2 = WMMA_IU8(a.v, b2.v, acc2);
        acc3 = WMMA_IU8(a.v, b3.v, acc3);
        buf ^= 1;
    }
    int ncol = (tn4 << 6) + (lane & 15);
    int mh = (lane < 16) ? 0 : 8;
    float bv0 = bias[ncol], bv1 = bias[ncol + 16];
    float bv2 = bias[ncol + 32], bv3 = bias[ncol + 48];
    for (int r = 0; r < 8; ++r) {
        long row = (long)(m0 + r + mh) * N;
        out[row + ncol]      = (float)acc0[r] + bv0;
        out[row + ncol + 16] = (float)acc1[r] + bv1;
        out[row + ncol + 32] = (float)acc2[r] + bv2;
        out[row + ncol + 48] = (float)acc3[r] + bv3;
    }
}

__global__ void maxpool_nhwc(const float* __restrict__ in, float* __restrict__ out,
                             int NB, int H, int W, int C)
{
    int Ho = H >> 1, Wo = W >> 1;
    int idx = blockIdx.x * blockDim.x + threadIdx.x;
    int total = NB * Ho * Wo * C;
    if (idx >= total) return;
    int c = idx % C;
    int t = idx / C;
    int wo = t % Wo; t /= Wo;
    int ho = t % Ho;
    int n = t / Ho;
    long base = ((long)(n * H + ho * 2) * W + wo * 2) * C + c;
    float a = in[base], b = in[base + C];
    float d = in[base + (long)W * C], e = in[base + (long)W * C + C];
    out[idx] = fmaxf(fmaxf(a, b), fmaxf(d, e));
}

__global__ void bn_stats(const float* __restrict__ x, int M, int C,
                         float* __restrict__ mean, float* __restrict__ rstd)
{
    __shared__ float ss[256], s2[256];
    int c = blockIdx.x;
    float a = 0.f, b = 0.f;
    for (int i = threadIdx.x; i < M; i += blockDim.x) {
        float v = x[(long)i * C + c];
        a += v; b += v * v;
    }
    ss[threadIdx.x] = a; s2[threadIdx.x] = b;
    __syncthreads();
    for (int s = 128; s > 0; s >>= 1) {
        if ((int)threadIdx.x < s) {
            ss[threadIdx.x] += ss[threadIdx.x + s];
            s2[threadIdx.x] += s2[threadIdx.x + s];
        }
        __syncthreads();
    }
    if (threadIdx.x == 0) {
        float m = ss[0] / (float)M;
        float v = s2[0] / (float)M - m * m;
        mean[c] = m;
        rstd[c] = rsqrtf(v + EPSV);
    }
}

__global__ void bn_apply(const float* __restrict__ x,
                         const float* __restrict__ mean, const float* __restrict__ rstd,
                         const float* __restrict__ g, const float* __restrict__ be,
                         int M, int C,
                         signed char* __restrict__ outb, float* __restrict__ outf)
{
    int idx = blockIdx.x * blockDim.x + threadIdx.x;
    if (idx >= M * C) return;
    int c = idx % C;
    float y = (x[idx] - mean[c]) * rstd[c] * g[c] + be[c];
    if (outb) outb[idx] = (y > 0.f) ? (signed char)1 : (signed char)0;
    if (outf) outf[idx] = (y > 0.f) ? y : 0.f;
}

__global__ void fc8_kernel(const float* __restrict__ h, const float* __restrict__ w8,
                           const float* __restrict__ b8, float* __restrict__ out)
{
    int idx = blockIdx.x * blockDim.x + threadIdx.x;
    if (idx >= 128 * 10) return;
    int j = idx % 10, n = idx / 10;
    float acc = b8[j];
    for (int k = 0; k < 1024; ++k)
        acc += h[n * 1024 + k] * w8[j * 1024 + k];
    out[idx] = acc;
}

__global__ void logsoftmax10(const float* __restrict__ z, float* __restrict__ out)
{
    int n = blockIdx.x * blockDim.x + threadIdx.x;
    if (n >= 128) return;
    float mx = -1e30f;
    for (int j = 0; j < 10; ++j) mx = fmaxf(mx, z[n * 10 + j]);
    float s = 0.f;
    for (int j = 0; j < 10; ++j) s += expf(z[n * 10 + j] - mx);
    float l = logf(s);
    for (int j = 0; j < 10; ++j) out[n * 10 + j] = z[n * 10 + j] - mx - l;
}

extern "C" void kernel_launch(void* const* d_in, const int* in_sizes, int n_in,
                              void* d_out, int out_size, void* d_ws, size_t ws_size,
                              hipStream_t stream)
{
    const int NB = 128;
    const float* x = (const float*)d_in[0];
    auto Wp  = [&](int i) { return (const float*)d_in[1 + 4 * i]; };
    auto Bp  = [&](int i) { return (const float*)d_in[2 + 4 * i]; };
    auto Gp  = [&](int i) { return (const float*)d_in[3 + 4 * i]; };
    auto BEp = [&](int i) { return (const float*)d_in[4 + 4 * i]; };
    const float* w8 = (const float*)d_in[33];
    const float* b8 = (const float*)d_in[34];

    char* ws = (char*)d_ws;
    size_t off = 0;
    auto alloc = [&](size_t bytes) -> char* {
        char* p = ws + off;
        off += (bytes + 255) & ~(size_t)255;
        return p;
    };

    float* F1 = (float*)alloc(67108864);
    float* F2 = (float*)alloc(16777216);
    signed char* I1 = (signed char*)alloc(16777216);
    signed char* I2 = (signed char*)alloc(16777216);
    signed char* WP1 = (signed char*)alloc(147456);
    signed char* WP2 = (signed char*)alloc(294912);
    signed char* WP3 = (signed char*)alloc(589824);
    signed char* WP4 = (signed char*)alloc(1179648);
    signed char* WP5 = (signed char*)alloc(2359296);
    signed char* WP6 = (signed char*)alloc(8388608);
    signed char* WP7 = (signed char*)alloc(1048576);
    float* MEAN = (float*)alloc(4096);
    float* RSTD = (float*)alloc(4096);
    float* RBUF = (float*)alloc(524288);
    float* LG   = (float*)alloc(5120);

    const int T = 256;
    auto blks = [&](long n) { return (unsigned)((n + T - 1) / T); };

    pack_conv_w<<<blks(9L * 2 * 8 * 1024), T, 0, stream>>>(Wp(1), WP1, 128, 128);
    pack_conv_w<<<blks(9L * 2 * 16 * 1024), T, 0, stream>>>(Wp(2), WP2, 128, 256);
    pack_conv_w<<<blks(9L * 4 * 16 * 1024), T, 0, stream>>>(Wp(3), WP3, 256, 256);
    pack_conv_w<<<blks(9L * 4 * 32 * 1024), T, 0, stream>>>(Wp(4), WP4, 256, 512);
    pack_conv_w<<<blks(9L * 8 * 32 * 1024), T, 0, stream>>>(Wp(5), WP5, 512, 512);
    pack_fc_w<<<blks(128L * 64 * 1024), T, 0, stream>>>(Wp(6), WP6, 8192, 1024, 1);
    pack_fc_w<<<blks(16L * 64 * 1024), T, 0, stream>>>(Wp(7), WP7, 1024, 1024, 0);

    conv0_direct<<<blks(NB * 32L * 32 * 128), T, 0, stream>>>(x, Wp(0), Bp(0), F1, NB);
    bn_stats<<<128, T, 0, stream>>>(F1, NB * 1024, 128, MEAN, RSTD);
    bn_apply<<<blks(NB * 1024L * 128), T, 0, stream>>>(F1, MEAN, RSTD, Gp(0), BEp(0),
                                                       NB * 1024, 128, I1, nullptr);

    bgemm_conv<<<(8192 / 8) * 2, T, 0, stream>>>(I1, WP1, Bp(1), F1,
                                                 NB, 32, 32, 128, 128);
    maxpool_nhwc<<<blks(NB * 256L * 128), T, 0, stream>>>(F1, F2, NB, 32, 32, 128);
    bn_stats<<<128, T, 0, stream>>>(F2, NB * 256, 128, MEAN, RSTD);
    bn_apply<<<blks(NB * 256L * 128), T, 0, stream>>>(F2, MEAN, RSTD, Gp(1), BEp(1),
                                                      NB * 256, 128, I2, nullptr);

    bgemm_conv<<<(2048 / 8) * 4, T, 0, stream>>>(I2, WP2, Bp(2), F1,
                                                 NB, 16, 16, 128, 256);
    bn_stats<<<256, T, 0, stream>>>(F1, NB * 256, 256, MEAN, RSTD);
    bn_apply<<<blks(NB * 256L * 256), T, 0, stream>>>(F1, MEAN, RSTD, Gp(2), BEp(2),
                                                      NB * 256, 256, I1, nullptr);

    bgemm_conv<<<(2048 / 8) * 4, T, 0, stream>>>(I1, WP3, Bp(3), F1,
                                                 NB, 16, 16, 256, 256);
    maxpool_nhwc<<<blks(NB * 64L * 256), T, 0, stream>>>(F1, F2, NB, 16, 16, 256);
    bn_stats<<<256, T, 0, stream>>>(F2, NB * 64, 256, MEAN, RSTD);
    bn_apply<<<blks(NB * 64L * 256), T, 0, stream>>>(F2, MEAN, RSTD, Gp(3), BEp(3),
                                                     NB * 64, 256, I2, nullptr);

    bgemm_conv<<<(512 / 8) * 8, T, 0, stream>>>(I2, WP4, Bp(4), F1,
                                                NB, 8, 8, 256, 512);
    bn_stats<<<512, T, 0, stream>>>(F1, NB * 64, 512, MEAN, RSTD);
    bn_apply<<<blks(NB * 64L * 512), T, 0, stream>>>(F1, MEAN, RSTD, Gp(4), BEp(4),
                                                     NB * 64, 512, I1, nullptr);

    bgemm_conv<<<(512 / 8) * 8, T, 0, stream>>>(I1, WP5, Bp(5), F1,
                                                NB, 8, 8, 512, 512);
    maxpool_nhwc<<<blks(NB * 16L * 512), T, 0, stream>>>(F1, F2, NB, 8, 8, 512);
    bn_stats<<<512, T, 0, stream>>>(F2, NB * 16, 512, MEAN, RSTD);
    bn_apply<<<blks(NB * 16L * 512), T, 0, stream>>>(F2, MEAN, RSTD, Gp(5), BEp(5),
                                                     NB * 16, 512, I2, nullptr);

    bgemm_fc<<<16, T, 0, stream>>>(I2, WP6, Bp(6), F1, 8192, 1024);
    bn_stats<<<1024, T, 0, stream>>>(F1, 128, 1024, MEAN, RSTD);
    bn_apply<<<blks(128L * 1024), T, 0, stream>>>(F1, MEAN, RSTD, Gp(6), BEp(6),
                                                  128, 1024, I1, nullptr);

    bgemm_fc<<<16, T, 0, stream>>>(I1, WP7, Bp(7), F2, 1024, 1024);
    bn_stats<<<1024, T, 0, stream>>>(F2, 128, 1024, MEAN, RSTD);
    bn_apply<<<blks(128L * 1024), T, 0, stream>>>(F2, MEAN, RSTD, Gp(7), BEp(7),
                                                  128, 1024, nullptr, RBUF);

    fc8_kernel<<<blks(1280), T, 0, stream>>>(RBUF, w8, b8, LG);
    logsoftmax10<<<1, 128, 0, stream>>>(LG, (float*)d_out);

    (void)in_sizes; (void)n_in; (void)out_size; (void)ws_size;
}
